// SemRelAttention_75041668596129
// MI455X (gfx1250) — compile-verified, matrix operand order checked by reading, not run on hardware
//
#include <hip/hip_runtime.h>
#include <hip/hip_bf16.h>

// ---------------------------------------------------------------------------
// SemRelAttention for MI455X (gfx1250, wave32, WMMA bf16 16x16x32, f32 acc)
// ---------------------------------------------------------------------------

typedef __attribute__((ext_vector_type(8)))  __bf16 v8bf;
typedef __attribute__((ext_vector_type(16))) __bf16 v16bf;
typedef __attribute__((ext_vector_type(8)))  float  v8f;

#define B_   4
#define C_   256
#define H_   128
#define W_   128
#define HW_  16384
#define NH_  4
#define DH_  64
#define SS_  256    // 16x16 pooled keys
#define INNER_ 256

__device__ __forceinline__ v8f vzero8() {
  v8f z;
#pragma unroll
  for (int i = 0; i < 8; ++i) z[i] = 0.0f;
  return z;
}

// Load one 16x16x32-bf16 A/B fragment row: `rowbase` points at this lane's
// row (A) / column (B), elements contiguous along K with leading dim >= kb+32.
// Lanes 0-15 hold K = kb+{0..7, 16..23}; lanes 16-31 hold K = kb+{8..15, 24..31}.
__device__ __forceinline__ v16bf ldfrag(const __bf16* rowbase, int kb, int lane) {
  const __bf16* p = rowbase + kb + ((lane & 16) ? 8 : 0);
  v8bf lo = *(const v8bf*)(p);
  v8bf hi = *(const v8bf*)(p + 16);
  return __builtin_shufflevector(lo, hi, 0, 1, 2, 3, 4, 5, 6, 7,
                                 8, 9, 10, 11, 12, 13, 14, 15);
}

__device__ __forceinline__ v8f wmma_bf16(v16bf a, v16bf b, v8f c) {
  return __builtin_amdgcn_wmma_f32_16x16x32_bf16(
      /*neg_a=*/false, a, /*neg_b=*/false, b,
      /*c_mod=*/(short)0, c, /*reuse_a=*/false, /*reuse_b=*/false);
}

__device__ __forceinline__ void atomicMaxF(float* addr, float v) {
  unsigned int* u = (unsigned int*)addr;
  unsigned int old = __atomic_load_n(u, __ATOMIC_RELAXED);
  while (true) {
    float f = __uint_as_float(old);
    if (f >= v) break;
    unsigned int assumed = old;
    old = atomicCAS(u, assumed, __float_as_uint(v));
    if (old == assumed) break;
  }
}

// ---------------------------------------------------------------------------
// K0: weight conversion f32 -> bf16, rel tables padded to 32 rows, pool init
// ---------------------------------------------------------------------------
__global__ void k_prep(const float* __restrict__ pw1, const float* __restrict__ pw2,
                       const float* __restrict__ krw, const float* __restrict__ krh,
                       __bf16* __restrict__ w1b, __bf16* __restrict__ w2b,
                       __bf16* __restrict__ krwb, __bf16* __restrict__ krhb,
                       float* __restrict__ kpool, float* __restrict__ vpool) {
  int i = blockIdx.x * blockDim.x + threadIdx.x;
  if (i < 768 * 256) w1b[i] = (__bf16)pw1[i];
  if (i < 256 * 256) w2b[i] = (__bf16)pw2[i];
  if (i < 32 * 64) {
    int r = i >> 6;
    float vw = (r < 31) ? krw[i] : 0.0f;
    float vh = (r < 31) ? krh[i] : 0.0f;
    krwb[i] = (__bf16)vw;
    krhb[i] = (__bf16)vh;
  }
  if (i < B_ * NH_ * SS_ * DH_) {
    kpool[i] = -3.0e38f;
    vpool[i] = -3.0e38f;
  }
}

// ---------------------------------------------------------------------------
// K1: depthwise 3x3 (f32 in) -> y1t bf16 [b][pix][ch]  (transposed for GEMM B)
// ---------------------------------------------------------------------------
__global__ void k_dw1(const float* __restrict__ x, const float* __restrict__ dw,
                      __bf16* __restrict__ y1t) {
  int blk = blockIdx.x;          // b*C*64 + c*64 + h2
  int h2 = blk & 63;
  int c  = (blk >> 6) & 255;
  int b  = blk >> 14;
  int tid = threadIdx.x;
  int h = h2 * 2 + (tid >> 7);
  int w = tid & 127;
  float wgt[9];
#pragma unroll
  for (int i = 0; i < 9; ++i) wgt[i] = dw[c * 9 + i];
  const float* xp = x + ((size_t)(b * C_ + c)) * HW_;
  float s = 0.0f;
#pragma unroll
  for (int dh = -1; dh <= 1; ++dh) {
    int hh = h + dh;
    if (hh < 0 || hh >= H_) continue;
#pragma unroll
    for (int dww = -1; dww <= 1; ++dww) {
      int ww = w + dww;
      if (ww < 0 || ww >= W_) continue;
      s += wgt[(dh + 1) * 3 + (dww + 1)] * xp[hh * W_ + ww];
    }
  }
  y1t[((size_t)b * HW_ + h * W_ + w) * C_ + c] = (__bf16)s;
}

// ---------------------------------------------------------------------------
// K2: pointwise-1 GEMM  qkv[768][HW] = W1[768][256] x Y1[256][HW]  (per batch)
//     epilogue: q -> head-major bf16; k,v -> fused 8x8 max-pool (atomic max)
// ---------------------------------------------------------------------------
__global__ __launch_bounds__(256) void k_pw1(const __bf16* __restrict__ w1b,
                                             const __bf16* __restrict__ y1t,
                                             __bf16* __restrict__ qb,
                                             float* __restrict__ kpool,
                                             float* __restrict__ vpool) {
  int lane = threadIdx.x & 31;
  int wv   = threadIdx.x >> 5;      // 0..7
  int nblk = blockIdx.x;            // 0..255 : 64-pixel block
  int mblk = blockIdx.y;            // 0..5   : 128 out-channels
  int b    = blockIdx.z;
  int m0 = mblk * 128 + wv * 16;
  int n0 = nblk * 64;
  const __bf16* A  = w1b + (size_t)(m0 + (lane & 15)) * 256;
  const __bf16* Bb = y1t + ((size_t)b * HW_ + n0) * 256;

  v8f acc[4];
#pragma unroll
  for (int t = 0; t < 4; ++t) acc[t] = vzero8();

  for (int kb = 0; kb < 256; kb += 32) {
    v16bf a = ldfrag(A, kb, lane);
#pragma unroll
    for (int t = 0; t < 4; ++t) {
      v16bf bb = ldfrag(Bb + (size_t)(t * 16 + (lane & 15)) * 256, kb, lane);
      acc[t] = wmma_bf16(a, bb, acc[t]);
    }
  }

  int mh = (lane >> 4) << 3;        // +8 rows for upper half-wave
#pragma unroll
  for (int t = 0; t < 4; ++t) {
    int pix = n0 + t * 16 + (lane & 15);
    int py = pix >> 7;              // image row  h
    int px = pix & 127;             // image col  w
    int key = ((py >> 3) << 4) + (px >> 3);   // ph*16 + pw
#pragma unroll
    for (int g = 0; g < 8; ++g) {
      int cch = m0 + g + mh;
      float v = acc[t][g];
      if (cch < 256) {               // q : channel = dh*4 + head
        int head = cch & 3, dh = cch >> 2;
        qb[(((size_t)b * NH_ + head) * HW_ + pix) * DH_ + dh] = (__bf16)v;
      } else if (cch < 512) {        // k : fused max-pool
        int ck = cch - 256, head = ck & 3, dh = ck >> 2;
        atomicMaxF(&kpool[(((size_t)b * NH_ + head) * SS_ + key) * DH_ + dh], v);
      } else {                       // v : fused max-pool
        int cv = cch - 512, head = cv & 3, dh = cv >> 2;
        atomicMaxF(&vpool[(((size_t)b * NH_ + head) * SS_ + key) * DH_ + dh], v);
      }
    }
  }
}

// ---------------------------------------------------------------------------
// K3: pooled k -> bf16 [bh][key][64];  pooled v -> bf16 transposed [bh][64][key]
// ---------------------------------------------------------------------------
__global__ void k_poolfin(const float* __restrict__ kpool, const float* __restrict__ vpool,
                          __bf16* __restrict__ kb16, __bf16* __restrict__ vtb) {
  int i = blockIdx.x * blockDim.x + threadIdx.x;   // over B*NH*256*64
  if (i >= B_ * NH_ * SS_ * DH_) return;
  kb16[i] = (__bf16)kpool[i];
  int bh  = i >> 14;
  int rem = i & 16383;
  int key = rem >> 6;
  int d   = rem & 63;
  vtb[(size_t)bh * (SS_ * DH_) + d * SS_ + key] = (__bf16)vpool[i];
}

// ---------------------------------------------------------------------------
// K4: fused attention: qk WMMA + factored rel-pos bias + softmax + attn store
//     + P*V WMMA.  One WG = one image row (128 pixels); one wave = 16 pixels
//     vs all 256 keys (16 C-tiles held in registers through softmax).
// ---------------------------------------------------------------------------
__global__ __launch_bounds__(256) void k_attn(const __bf16* __restrict__ qb,
                                              const __bf16* __restrict__ kb16,
                                              const __bf16* __restrict__ vtb,
                                              const __bf16* __restrict__ krwb,
                                              const __bf16* __restrict__ krhb,
                                              float* __restrict__ attn_out,
                                              __bf16* __restrict__ ob16) {
  __shared__ float  lhs[8][16][32];   // rel-h logits per wave, rows x (2S-1 pad 32)
  __shared__ float  lws[8][16][32];   // rel-w logits
  __shared__ __bf16 ps [8][16][32];   // P staging (C-layout -> A-layout)

  int lane = threadIdx.x & 31;
  int wv   = threadIdx.x >> 5;        // 0..7
  int x    = blockIdx.x;              // image row 0..127
  int head = blockIdx.y;
  int b    = blockIdx.z;
  size_t bh = (size_t)b * NH_ + head;
  int y0 = wv * 16;
  int mh = (lane >> 4) << 3;
  int nl = lane & 15;

  // ---- q A-fragments (16 rows x 64) ----
  const __bf16* Q = qb + (bh * HW_ + x * W_ + y0) * DH_;
  v16bf a0 = ldfrag(Q + (size_t)nl * DH_, 0,  lane);
  v16bf a1 = ldfrag(Q + (size_t)nl * DH_, 32, lane);

  // ---- qk: 16 key tiles x (K=64 -> 2 wmma) ----
  const __bf16* K = kb16 + bh * (SS_ * DH_);
  v8f acc[16];
#pragma unroll
  for (int t = 0; t < 16; ++t) acc[t] = vzero8();
#pragma unroll
  for (int t = 0; t < 16; ++t) {
    const __bf16* kr = K + (size_t)(t * 16 + nl) * DH_;
    v16bf b0 = ldfrag(kr, 0,  lane);
    v16bf b1 = ldfrag(kr, 32, lane);
    acc[t] = wmma_bf16(a0, b0, acc[t]);
    acc[t] = wmma_bf16(a1, b1, acc[t]);
  }

  // ---- rel-pos logits: Lh, Lw = q x key_rel^T (64 x 32-padded) ----
  v8f lh[2], lw[2];
#pragma unroll
  for (int t = 0; t < 2; ++t) { lh[t] = vzero8(); lw[t] = vzero8(); }
#pragma unroll
  for (int t = 0; t < 2; ++t) {
    const __bf16* rh = krhb + (size_t)(t * 16 + nl) * DH_;
    const __bf16* rw = krwb + (size_t)(t * 16 + nl) * DH_;
    lh[t] = wmma_bf16(a0, ldfrag(rh, 0, lane),  lh[t]);
    lh[t] = wmma_bf16(a1, ldfrag(rh, 32, lane), lh[t]);
    lw[t] = wmma_bf16(a0, ldfrag(rw, 0, lane),  lw[t]);
    lw[t] = wmma_bf16(a1, ldfrag(rw, 32, lane), lw[t]);
  }
#pragma unroll
  for (int t = 0; t < 2; ++t) {
    int r = t * 16 + nl;
#pragma unroll
    for (int g = 0; g < 8; ++g) {
      int m = g + mh;
      lhs[wv][m][r] = lh[t][g];
      lws[wv][m][r] = lw[t][g];
    }
  }
  // wave-private LDS region; DS ops from one wave complete in order.

  // ---- bias gather + scale ----
  const float scale = 0.125f;         // 64^-0.5
  int xh = x >> 3;
#pragma unroll
  for (int t = 0; t < 16; ++t) {      // hk = t, wk = nl
    int ih = t - xh + 15;             // in [0,30]
#pragma unroll
    for (int g = 0; g < 8; ++g) {
      int m = g + mh;
      int yy = y0 + m;
      int iw = nl - (yy >> 3) + 15;   // in [0,30]
      acc[t][g] = (acc[t][g] + lhs[wv][m][ih] + lws[wv][m][iw]) * scale;
    }
  }

  // ---- softmax across 256 keys (16 tiles x 16 lanes within half-wave) ----
  float mx[8], sm[8];
#pragma unroll
  for (int g = 0; g < 8; ++g) {
    mx[g] = -3.0e38f;
#pragma unroll
    for (int t = 0; t < 16; ++t) mx[g] = fmaxf(mx[g], acc[t][g]);
  }
#pragma unroll
  for (int d = 1; d < 16; d <<= 1)
#pragma unroll
    for (int g = 0; g < 8; ++g) mx[g] = fmaxf(mx[g], __shfl_xor(mx[g], d, 32));
#pragma unroll
  for (int g = 0; g < 8; ++g) sm[g] = 0.0f;
#pragma unroll
  for (int t = 0; t < 16; ++t)
#pragma unroll
    for (int g = 0; g < 8; ++g) {
      float e = __expf(acc[t][g] - mx[g]);
      acc[t][g] = e;
      sm[g] += e;
    }
#pragma unroll
  for (int d = 1; d < 16; d <<= 1)
#pragma unroll
    for (int g = 0; g < 8; ++g) sm[g] += __shfl_xor(sm[g], d, 32);
#pragma unroll
  for (int g = 0; g < 8; ++g) sm[g] = 1.0f / sm[g];
#pragma unroll
  for (int t = 0; t < 16; ++t)
#pragma unroll
    for (int g = 0; g < 8; ++g) acc[t][g] *= sm[g];

  // ---- attn (f32) is an output ----
  float* arow = attn_out + (bh * HW_ + x * W_ + y0) * SS_;
#pragma unroll
  for (int t = 0; t < 16; ++t)
#pragma unroll
    for (int g = 0; g < 8; ++g)
      arow[(size_t)(g + mh) * SS_ + t * 16 + nl] = acc[t][g];

  // ---- out = P x V  (K=256: 8 steps; re-stage P through LDS as A-frags) ----
  const __bf16* V = vtb + bh * (SS_ * DH_);   // [64][256]
  v8f oa[4];
#pragma unroll
  for (int vt = 0; vt < 4; ++vt) oa[vt] = vzero8();
#pragma unroll
  for (int kk = 0; kk < 8; ++kk) {
#pragma unroll
    for (int tt = 0; tt < 2; ++tt) {
      int t = kk * 2 + tt;
#pragma unroll
      for (int g = 0; g < 8; ++g)
        ps[wv][g + mh][tt * 16 + nl] = (__bf16)acc[t][g];
    }
    v16bf pa = ldfrag(&ps[wv][nl][0], 0, lane);
#pragma unroll
    for (int vt = 0; vt < 4; ++vt) {
      v16bf vb = ldfrag(V + (size_t)(vt * 16 + nl) * SS_, kk * 32, lane);
      oa[vt] = wmma_bf16(pa, vb, oa[vt]);
    }
  }

  // ---- scatter to NCHW bf16, channel = d*4 + head ----
#pragma unroll
  for (int vt = 0; vt < 4; ++vt) {
#pragma unroll
    for (int g = 0; g < 8; ++g) {
      int d = vt * 16 + nl;
      int cch = d * NH_ + head;
      int pix = x * W_ + y0 + g + mh;
      ob16[((size_t)b * INNER_ + cch) * HW_ + pix] = (__bf16)oa[vt][g];
    }
  }
}

// ---------------------------------------------------------------------------
// K5: depthwise 3x3 (bf16 in) -> y2t bf16 [b][pix][ch]
// ---------------------------------------------------------------------------
__global__ void k_dw2(const __bf16* __restrict__ o, const float* __restrict__ dw,
                      __bf16* __restrict__ y2t) {
  int blk = blockIdx.x;
  int h2 = blk & 63;
  int c  = (blk >> 6) & 255;
  int b  = blk >> 14;
  int tid = threadIdx.x;
  int h = h2 * 2 + (tid >> 7);
  int w = tid & 127;
  float wgt[9];
#pragma unroll
  for (int i = 0; i < 9; ++i) wgt[i] = dw[c * 9 + i];
  const __bf16* op = o + ((size_t)(b * INNER_ + c)) * HW_;
  float s = 0.0f;
#pragma unroll
  for (int dh = -1; dh <= 1; ++dh) {
    int hh = h + dh;
    if (hh < 0 || hh >= H_) continue;
#pragma unroll
    for (int dww = -1; dww <= 1; ++dww) {
      int ww = w + dww;
      if (ww < 0 || ww >= W_) continue;
      s += wgt[(dh + 1) * 3 + (dww + 1)] * (float)op[hh * W_ + ww];
    }
  }
  y2t[((size_t)b * HW_ + h * W_ + w) * INNER_ + c] = (__bf16)s;
}

// ---------------------------------------------------------------------------
// K6: pointwise-2 GEMM  out[256][HW] = W2[256][256] x Y2[256][HW], f32 store
// ---------------------------------------------------------------------------
__global__ __launch_bounds__(256) void k_pw2(const __bf16* __restrict__ w2b,
                                             const __bf16* __restrict__ y2t,
                                             float* __restrict__ out) {
  int lane = threadIdx.x & 31;
  int wv   = threadIdx.x >> 5;
  int nblk = blockIdx.x;            // 0..255
  int mblk = blockIdx.y;            // 0..1
  int b    = blockIdx.z;
  int m0 = mblk * 128 + wv * 16;
  int n0 = nblk * 64;
  const __bf16* A  = w2b + (size_t)(m0 + (lane & 15)) * 256;
  const __bf16* Bb = y2t + ((size_t)b * HW_ + n0) * 256;

  v8f acc[4];
#pragma unroll
  for (int t = 0; t < 4; ++t) acc[t] = vzero8();

  for (int kb = 0; kb < 256; kb += 32) {
    v16bf a = ldfrag(A, kb, lane);
#pragma unroll
    for (int t = 0; t < 4; ++t) {
      v16bf bb = ldfrag(Bb + (size_t)(t * 16 + (lane & 15)) * 256, kb, lane);
      acc[t] = wmma_bf16(a, bb, acc[t]);
    }
  }

  int mh = (lane >> 4) << 3;
#pragma unroll
  for (int t = 0; t < 4; ++t) {
    int pix = n0 + t * 16 + (lane & 15);
#pragma unroll
    for (int g = 0; g < 8; ++g) {
      int cch = m0 + g + mh;
      out[((size_t)b * C_ + cch) * HW_ + pix] = acc[t][g];
    }
  }
}

// ---------------------------------------------------------------------------
extern "C" void kernel_launch(void* const* d_in, const int* in_sizes, int n_in,
                              void* d_out, int out_size, void* d_ws, size_t ws_size,
                              hipStream_t stream) {
  const float* x    = (const float*)d_in[0];
  const float* dw1  = (const float*)d_in[1];
  const float* pw1  = (const float*)d_in[2];
  const float* krw  = (const float*)d_in[3];   // key_rel_w (31x64)
  const float* krh  = (const float*)d_in[4];   // key_rel_h (31x64)
  const float* dw2  = (const float*)d_in[5];
  const float* pw2  = (const float*)d_in[6];

  float* out_final = (float*)d_out;                              // (B,256,H,W)
  float* attn_out  = out_final + (size_t)B_ * C_ * HW_;          // (B,4,HW,256)

  size_t off = 0;
  char* wsc = (char*)d_ws;
  auto take = [&](size_t bytes) -> void* {
    void* p = wsc + off;
    off += (bytes + 255) & ~(size_t)255;
    return p;
  };
  __bf16* w1b  = (__bf16*)take((size_t)768 * 256 * 2);
  __bf16* w2b  = (__bf16*)take((size_t)256 * 256 * 2);
  __bf16* krwb = (__bf16*)take((size_t)32 * 64 * 2);
  __bf16* krhb = (__bf16*)take((size_t)32 * 64 * 2);
  float*  kpool = (float*)take((size_t)B_ * NH_ * SS_ * DH_ * 4);
  float*  vpool = (float*)take((size_t)B_ * NH_ * SS_ * DH_ * 4);
  __bf16* kb16 = (__bf16*)take((size_t)B_ * NH_ * SS_ * DH_ * 2);
  __bf16* vtb  = (__bf16*)take((size_t)B_ * NH_ * SS_ * DH_ * 2);
  __bf16* y1t  = (__bf16*)take((size_t)B_ * HW_ * C_ * 2);
  __bf16* qb   = (__bf16*)take((size_t)B_ * NH_ * HW_ * DH_ * 2);
  __bf16* ob   = (__bf16*)take((size_t)B_ * INNER_ * HW_ * 2);
  __bf16* y2t  = (__bf16*)take((size_t)B_ * HW_ * INNER_ * 2);
  (void)ws_size; (void)in_sizes; (void)n_in; (void)out_size;

  // K0: conversions + pool init (262144 max elements)
  k_prep<<<1024, 256, 0, stream>>>(pw1, pw2, krw, krh, w1b, w2b, krwb, krhb,
                                   kpool, vpool);
  // K1: depthwise-1
  k_dw1<<<B_ * C_ * 64, 256, 0, stream>>>(x, dw1, y1t);
  // K2: pointwise-1 GEMM + fused pooling
  {
    dim3 g(256, 6, B_);
    k_pw1<<<g, 256, 0, stream>>>(w1b, y1t, qb, kpool, vpool);
  }
  // K3: pool finalize
  k_poolfin<<<1024, 256, 0, stream>>>(kpool, vpool, kb16, vtb);
  // K4: fused attention
  {
    dim3 g(H_, NH_, B_);
    k_attn<<<g, 256, 0, stream>>>(qb, kb16, vtb, krwb, krhb, attn_out, ob);
  }
  // K5: depthwise-2
  k_dw2<<<B_ * INNER_ * 64, 256, 0, stream>>>(ob, dw2, y2t);
  // K6: pointwise-2 GEMM -> final output
  {
    dim3 g(256, 2, B_);
    k_pw2<<<g, 256, 0, stream>>>(w2b, y2t, out_final);
  }
}
